// VQGAN_91302414778646
// MI455X (gfx1250) — matrix operand order repaired and checked by reading, not compiled, not run on hardware
//
#include <hip/hip_runtime.h>
#include <hip/hip_bf16.h>
#include <math.h>

// ---- vector types (avoid HIP's predefined ushortN names) ----
typedef __attribute__((ext_vector_type(16))) __bf16        v16bf;
typedef __attribute__((ext_vector_type(8)))  float         v8f;
typedef __attribute__((ext_vector_type(8)))  unsigned short us8;

__device__ __forceinline__ unsigned short f2bf(float f) {
    union { float f; unsigned u; } v; v.f = f;
    unsigned r = v.u + 0x7FFFu + ((v.u >> 16) & 1u);   // round-to-nearest-even
    return (unsigned short)(r >> 16);
}
__device__ __forceinline__ float bf2f(unsigned short h) {
    union { unsigned u; float f; } v; v.u = ((unsigned)h) << 16; return v.f;
}

// ===================== weight repack into WMMA B-fragment order ==============
// B-fragment layout (bf16 32x16 tile): index = ((kt*NT + nt)*32 + lane)*16 + t
//   col = nt*16 + (lane&15),  k = kt*32 + (t>>3)*16 + (lane>>4)*8 + (t&7)
__global__ void __launch_bounds__(256)
pack_b_kernel(const float* __restrict__ W, unsigned short* __restrict__ Bf,
              int N, int mode, int total)
{
    int id = blockIdx.x * 256 + threadIdx.x;
    if (id >= total) return;
    int t    = id & 15;
    int lane = (id >> 4) & 31;
    int tile = id >> 9;
    int NT   = N >> 4;
    int kt = tile / NT, nt = tile % NT;
    int col = nt * 16 + (lane & 15);
    int k   = kt * 32 + (t >> 3) * 16 + (lane >> 4) * 8 + (t & 7);
    float val;
    switch (mode) {
        case 1: val = (k < 16) ? W[col * 16 + k] : 0.f; break;                 // w1 [64,1,4,4], K padded 16->32
        case 2: { int s = k >> 6, i = k & 63; val = W[col * 256 + i * 4 + s]; } break; // w2 [128,64,2,2]
        case 3: val = W[col * 128 + k]; break;                                 // w3 [256,128,1,1]
        case 4: { int s = col >> 7, o = col & 127; val = W[k * 512 + o * 4 + s]; } break;  // wt1 [256,128,2,2]
        case 5: { int s = col >> 6, o = col & 63;  val = W[k * 1024 + o * 16 + s]; } break; // wt2 [128,64,4,4]
        default: val = W[col * 256 + k]; break;                                // codebook [1024,256] -> B[256,1024]
    }
    Bf[id] = f2bf(val);
}

__global__ void c2_kernel(const float* __restrict__ cb, float* __restrict__ c2)
{
    int v = blockIdx.x * 256 + threadIdx.x;
    if (v >= 1024) return;
    float s = 0.f;
    for (int e = 0; e < 256; ++e) { float x = cb[v * 256 + e]; s += x * x; }
    c2[v] = s;
}

// ===================== conv1 im2col (4x4 patches, K padded to 32) ============
__global__ void __launch_bounds__(256)
im2col1_kernel(const float* __restrict__ x, unsigned short* __restrict__ A1)
{
    int id = blockIdx.x * 256 + threadIdx.x;    // 262144*32 elements
    int kk = id & 31, n = id >> 5;
    float v = 0.f;
    if (kk < 16) {
        int ky = kk >> 2, kx = kk & 3;
        int b = n >> 12, rem = n & 4095, y = rem >> 6, xx = rem & 63;
        v = x[(size_t)b * 65536 + (size_t)(y * 4 + ky) * 256 + (xx * 4 + kx)];
    }
    A1[id] = f2bf(v);
}

// ===================== generic bf16 WMMA GEMM with fused epilogue ============
// A: row-major [M,K] bf16.  Bf: fragment-packed [K,N].  8 waves/block; each
// wave owns a 16(M) x 64(N) strip: one A fragment feeds 4 WMMAs per K-step
// (2 A-loads + 8 B-loads per 4 WMMAs).  Block tile = 128(M) x 64(N).
// MODE 0: conv1 -> h1 scattered into conv2 im2col layout, +bias,ReLU
// MODE 1: plain row-major bf16 out, +bias,ReLU                (conv2)
// MODE 2: z_e: f32 NCHW to d_out AND bf16 row-major, +bias    (conv3)
// MODE 3: deconv1 sub-pixel scatter to d1 NHWC, +bias,ReLU
// MODE 4: deconv2 sub-pixel scatter to d2 NHWC, +bias,ReLU
template<int MODE>
__global__ void __launch_bounds__(256)
gemm_wmma_bf16(const unsigned short* __restrict__ A,
               const unsigned short* __restrict__ Bf,
               const float* __restrict__ bias,
               int M, int N, int K,
               float* __restrict__ outF,
               unsigned short* __restrict__ outH)
{
    const int lane   = threadIdx.x & 31;
    const int wave   = threadIdx.x >> 5;
    const int row0   = (blockIdx.x * 8 + wave) << 4;
    const int ntBase = blockIdx.y * 4;
    const int g      = lane >> 4;
    const int NT     = N >> 4;

    union Frag { us8 h[2]; v16bf v; };

    // A 16x32 bf16 fragment: lane m=(lane&15); g picks K sub-halves {0..7,16..23}/{8..15,24..31}
    const unsigned short* aPtr = A + (size_t)(row0 + (lane & 15)) * K + g * 8;
    const unsigned short* bPtr = Bf + ((size_t)ntBase * 32 + lane) * 16;
    const size_t bStep = (size_t)NT * 512;     // one K-step (32 rows of B)

    v8f acc[4];
    #pragma unroll
    for (int tt = 0; tt < 4; ++tt) acc[tt] = (v8f){0.f,0.f,0.f,0.f,0.f,0.f,0.f,0.f};

    const int KT = K >> 5;
    for (int kt = 0; kt < KT; ++kt) {
        Frag a;
        a.h[0] = *reinterpret_cast<const us8*>(aPtr);
        a.h[1] = *reinterpret_cast<const us8*>(aPtr + 16);
        #pragma unroll
        for (int tt = 0; tt < 4; ++tt) {
            Frag b;
            b.h[0] = *reinterpret_cast<const us8*>(bPtr + tt * 512);
            b.h[1] = *reinterpret_cast<const us8*>(bPtr + tt * 512 + 8);
            acc[tt] = __builtin_amdgcn_wmma_f32_16x16x32_bf16(false, a.v, false, b.v,
                                                              (short)0, acc[tt], false, false);
        }
        aPtr += 32;
        bPtr += bStep;
    }

    #pragma unroll
    for (int tt = 0; tt < 4; ++tt) {
        const int col = (ntBase + tt) * 16 + (lane & 15);
        const float bc = bias[col];
        #pragma unroll
        for (int j = 0; j < 8; ++j) {
            int n = row0 + j + (g << 3);      // C/D layout: lanes16-31 hold M=j+8
            float v = acc[tt][j] + bc;
            if (MODE != 2) v = fmaxf(v, 0.f);
            if (MODE == 0) {
                int b_ = n >> 12, rem = n & 4095, y = rem >> 6, x = rem & 63;
                size_t dst = (((size_t)b_ * 32 + (y >> 1)) * 32 + (x >> 1)) * 256
                           + (((y & 1) * 2 + (x & 1)) * 64) + col;
                outH[dst] = f2bf(v);
            } else if (MODE == 1) {
                outH[(size_t)n * N + col] = f2bf(v);
            } else if (MODE == 2) {
                int b_ = n >> 10, pix = n & 1023;
                outF[((size_t)b_ * 256 + col) * 1024 + pix] = v;   // z_e NCHW f32
                outH[(size_t)n * 256 + col] = f2bf(v);             // z_e flat bf16
            } else if (MODE == 3) {
                int s = col >> 7, o = col & 127, p = s >> 1, q = s & 1;
                int b_ = n >> 10, pix = n & 1023, h = pix >> 5, w = pix & 31;
                outH[(((size_t)b_ * 64 + 2 * h + p) * 64 + (2 * w + q)) * 128 + o] = f2bf(v);
            } else {
                int s = col >> 6, o = col & 63, p = s >> 2, q = s & 3;
                int b_ = n >> 12, pix = n & 4095, y = pix >> 6, x = pix & 63;
                outH[(((size_t)b_ * 256 + 4 * y + p) * 256 + (4 * x + q)) * 64 + o] = f2bf(v);
            }
        }
    }
}

// ===================== fused distance GEMM + argmin ==========================
// One block handles 16 z rows against all 1024 codes: wave w sweeps col tiles
// {w, w+8, ...}; argmin of (|c|^2 - 2 z.c) == argmin of full distance.
// (Compiler hoists all 8 A fragments into persistent VGPRs -- full A reuse.)
__global__ void __launch_bounds__(256)
vq_argmin_kernel(const unsigned short* __restrict__ A,    // z_e [65536,256] bf16
                 const unsigned short* __restrict__ CBf,  // packed codebook B [256,1024]
                 const float* __restrict__ c2,
                 int* __restrict__ idxOut)
{
    const int lane = threadIdx.x & 31;
    const int wave = threadIdx.x >> 5;
    const int g    = lane >> 4;
    const int row0 = blockIdx.x << 4;
    const unsigned short* aBase = A + (size_t)(row0 + (lane & 15)) * 256 + g * 8;

    union Frag { us8 h[2]; v16bf v; };

    float bv[8]; int bi[8];
    #pragma unroll
    for (int j = 0; j < 8; ++j) { bv[j] = 3.4e38f; bi[j] = 0; }

    for (int ct = wave; ct < 64; ct += 8) {
        v8f acc = {0.f, 0.f, 0.f, 0.f, 0.f, 0.f, 0.f, 0.f};
        const unsigned short* aPtr = aBase;
        const unsigned short* bPtr = CBf + ((size_t)ct * 32 + lane) * 16;
        #pragma unroll
        for (int kt = 0; kt < 8; ++kt) {
            Frag a, b;
            a.h[0] = *reinterpret_cast<const us8*>(aPtr);
            a.h[1] = *reinterpret_cast<const us8*>(aPtr + 16);
            b.h[0] = *reinterpret_cast<const us8*>(bPtr);
            b.h[1] = *reinterpret_cast<const us8*>(bPtr + 8);
            acc = __builtin_amdgcn_wmma_f32_16x16x32_bf16(false, a.v, false, b.v,
                                                          (short)0, acc, false, false);
            aPtr += 32;
            bPtr += (size_t)64 * 512;
        }
        int c = ct * 16 + (lane & 15);
        float cc = c2[c];
        #pragma unroll
        for (int j = 0; j < 8; ++j) {
            float d = cc - 2.f * acc[j];
            if (d < bv[j] || (d == bv[j] && c < bi[j])) { bv[j] = d; bi[j] = c; }
        }
    }
    // reduce across the 16 lanes of each half-wave (same rows, different cols)
    #pragma unroll
    for (int m = 1; m < 16; m <<= 1) {
        #pragma unroll
        for (int j = 0; j < 8; ++j) {
            float ov = __shfl_xor(bv[j], m, 32);
            int   oi = __shfl_xor(bi[j], m, 32);
            if (ov < bv[j] || (ov == bv[j] && oi < bi[j])) { bv[j] = ov; bi[j] = oi; }
        }
    }
    __shared__ float sv[8][16];
    __shared__ int   si[8][16];
    if ((lane & 15) == 0) {
        #pragma unroll
        for (int j = 0; j < 8; ++j) { sv[wave][j + (g << 3)] = bv[j]; si[wave][j + (g << 3)] = bi[j]; }
    }
    __syncthreads();
    if (threadIdx.x < 16) {
        float best = sv[0][threadIdx.x]; int bidx = si[0][threadIdx.x];
        for (int w = 1; w < 8; ++w) {
            float v = sv[w][threadIdx.x]; int i = si[w][threadIdx.x];
            if (v < best || (v == best && i < bidx)) { best = v; bidx = i; }
        }
        idxOut[row0 + threadIdx.x] = bidx;
    }
}

// ===================== z_q gather (f32 NCHW out + bf16 decoder input) ========
__global__ void __launch_bounds__(256)
gather_zq_kernel(const int* __restrict__ idx, const float* __restrict__ cb,
                 float* __restrict__ zqF, unsigned short* __restrict__ A5)
{
    int id = blockIdx.x * 256 + threadIdx.x;    // 65536*256
    int e = id & 255, n = id >> 8;
    float v = cb[(size_t)idx[n] * 256 + e];
    int b = n >> 10, pix = n & 1023;
    zqF[((size_t)b * 256 + e) * 1024 + pix] = v;
    A5[id] = f2bf(v);
}

// ===================== final 3x3 conv (64->1) + tanh =========================
__global__ void __launch_bounds__(256)
final_conv_tanh_kernel(const unsigned short* __restrict__ d2,   // [B,256,256,64] bf16
                       const float* __restrict__ wd, const float* __restrict__ bd,
                       float* __restrict__ out)
{
    __shared__ float swd[576];
    for (int i = threadIdx.x; i < 576; i += 256) swd[i] = wd[i];
    __syncthreads();
    int id = blockIdx.x * 256 + threadIdx.x;    // 64*256*256
    int b = id >> 16, rem = id & 65535, Y = rem >> 8, X = rem & 255;
    float acc = bd[0];
    for (int dy = 0; dy < 3; ++dy) {
        int yy = Y + dy - 1;
        if (yy < 0 || yy > 255) continue;
        for (int dx = 0; dx < 3; ++dx) {
            int xx = X + dx - 1;
            if (xx < 0 || xx > 255) continue;
            const unsigned short* p = d2 + (((size_t)b * 256 + yy) * 256 + xx) * 64;
            const float* wp = swd + dy * 3 + dx;
            #pragma unroll
            for (int c8 = 0; c8 < 8; ++c8) {               // 8 x b128 channel loads
                us8 ch = *reinterpret_cast<const us8*>(p + c8 * 8);
                #pragma unroll
                for (int c = 0; c < 8; ++c)
                    acc += bf2f(ch[c]) * wp[(c8 * 8 + c) * 9];
            }
        }
    }
    out[id] = tanhf(acc);
}

// ===================== workspace layout (bytes) ==============================
// d2 (512MB) is written last among ws buffers; everything except d1/B5f is
// dead by then, so the early-phase buffers alias the d2 region.
// Required ws_size: 604,241,920 bytes (~576 MiB).
static constexpr size_t OFF_D1 = 0;                               // 67,108,864
static constexpr size_t OFF_B5 = 67108864;                        //    262,144
static constexpr size_t OFF_D2 = 67371008;                        // 536,870,912
static constexpr size_t OFF_A1 = 67371008;                        // 16,777,216 (aliases D2)
static constexpr size_t OFF_A2 = 84148224;                        // 33,554,432
static constexpr size_t OFF_A3 = 117702656;                       // 16,777,216
static constexpr size_t OFF_A4 = 134479872;                       // 33,554,432
static constexpr size_t OFF_A5 = 168034304;                       // 33,554,432
static constexpr size_t OFF_B1 = 201588736;                       //      4,096
static constexpr size_t OFF_B2 = 201592832;                       //     65,536
static constexpr size_t OFF_B3 = 201658368;                       //     65,536
static constexpr size_t OFF_B4 = 201723904;                       //    262,144
static constexpr size_t OFF_CB = 201986048;                       //    524,288
static constexpr size_t OFF_C2 = 202510336;                       //      4,096

extern "C" void kernel_launch(void* const* d_in, const int* in_sizes, int n_in,
                              void* d_out, int out_size, void* d_ws, size_t ws_size,
                              hipStream_t stream)
{
    (void)in_sizes; (void)n_in; (void)out_size; (void)ws_size;

    const float* x   = (const float*)d_in[0];
    const float* w1  = (const float*)d_in[1];
    const float* b1  = (const float*)d_in[2];
    const float* w2  = (const float*)d_in[3];
    const float* b2  = (const float*)d_in[4];
    const float* w3  = (const float*)d_in[5];
    const float* b3  = (const float*)d_in[6];
    const float* cb  = (const float*)d_in[7];
    const float* wt1 = (const float*)d_in[8];
    const float* bt1 = (const float*)d_in[9];
    const float* wt2 = (const float*)d_in[10];
    const float* bt2 = (const float*)d_in[11];
    const float* wd  = (const float*)d_in[12];
    const float* bd  = (const float*)d_in[13];

    // d_out: x_hat[4194304] f32 | indices[65536] i32 | z_e[16777216] f32 | z_q[16777216] f32
    float* xhat = (float*)d_out;
    int*   ind  = (int*)((float*)d_out + 4194304);
    float* zeF  = (float*)d_out + 4194304 + 65536;
    float* zqF  = zeF + 16777216;

    char* ws = (char*)d_ws;
    unsigned short* A1  = (unsigned short*)(ws + OFF_A1);
    unsigned short* A2  = (unsigned short*)(ws + OFF_A2);
    unsigned short* A3  = (unsigned short*)(ws + OFF_A3);
    unsigned short* A4  = (unsigned short*)(ws + OFF_A4);
    unsigned short* A5  = (unsigned short*)(ws + OFF_A5);
    unsigned short* B1f = (unsigned short*)(ws + OFF_B1);
    unsigned short* B2f = (unsigned short*)(ws + OFF_B2);
    unsigned short* B3f = (unsigned short*)(ws + OFF_B3);
    unsigned short* B4f = (unsigned short*)(ws + OFF_B4);
    unsigned short* B5f = (unsigned short*)(ws + OFF_B5);
    unsigned short* CBf = (unsigned short*)(ws + OFF_CB);
    float*          c2  = (float*)(ws + OFF_C2);
    unsigned short* D1  = (unsigned short*)(ws + OFF_D1);
    unsigned short* D2  = (unsigned short*)(ws + OFF_D2);

    // --- 1. repack weights to WMMA B-fragment order + |c|^2 ---
    pack_b_kernel<<<2048 / 256, 256, 0, stream>>>(w1,  B1f,   64, 1, 2048);
    pack_b_kernel<<<32768 / 256, 256, 0, stream>>>(w2,  B2f,  128, 2, 32768);
    pack_b_kernel<<<32768 / 256, 256, 0, stream>>>(w3,  B3f,  256, 3, 32768);
    pack_b_kernel<<<131072 / 256, 256, 0, stream>>>(wt1, B4f,  512, 4, 131072);
    pack_b_kernel<<<131072 / 256, 256, 0, stream>>>(wt2, B5f, 1024, 5, 131072);
    pack_b_kernel<<<262144 / 256, 256, 0, stream>>>(cb,  CBf, 1024, 6, 262144);
    c2_kernel<<<4, 256, 0, stream>>>(cb, c2);

    // --- 2. encoder ---  (grid = (M/128, N/64))
    im2col1_kernel<<<32768, 256, 0, stream>>>(x, A1);
    gemm_wmma_bf16<0><<<dim3(2048,  1), 256, 0, stream>>>(A1, B1f, b1, 262144, 64, 32, nullptr, A2);
    gemm_wmma_bf16<1><<<dim3(512,   2), 256, 0, stream>>>(A2, B2f, b2, 65536, 128, 256, nullptr, A3);
    gemm_wmma_bf16<2><<<dim3(512,   4), 256, 0, stream>>>(A3, B3f, b3, 65536, 256, 128, zeF, A4);

    // --- 3. vector quantization ---
    vq_argmin_kernel<<<4096, 256, 0, stream>>>(A4, CBf, c2, ind);
    gather_zq_kernel<<<65536, 256, 0, stream>>>(ind, cb, zqF, A5);

    // --- 4. decoder ---
    gemm_wmma_bf16<3><<<dim3(512,   8), 256, 0, stream>>>(A5, B4f, bt1, 65536, 512, 256, nullptr, D1);
    gemm_wmma_bf16<4><<<dim3(2048, 16), 256, 0, stream>>>(D1, B5f, bt2, 262144, 1024, 128, nullptr, D2);
    final_conv_tanh_kernel<<<16384, 256, 0, stream>>>(D2, wd, bd, xhat);
}
